// DADE_7026566496863
// MI455X (gfx1250) — matrix operand order repaired and checked by reading, not compiled, not run on hardware
//
#include <hip/hip_runtime.h>
#include <hip/hip_bf16.h>
#include <stdint.h>

typedef __attribute__((ext_vector_type(16))) __bf16 v16bf;
typedef __attribute__((ext_vector_type(8)))  __bf16 v8bf;
typedef __attribute__((ext_vector_type(8)))  float  v8f;

constexpr int kL  = 3;
constexpr int kN  = 8;
constexpr int kS  = 4;
constexpr int kT  = 1370;
constexpr int kD  = 768;
constexpr int kQ  = 1369;              // tokens per image after dropping CLS
constexpr int kSQ = kS * kQ;           // 5476 support tokens per image
constexpr int kNQROWS = kN * kQ;       // 10952 query rows
constexpr int kNSROWS = kN * kSQ;      // 43808 support rows
constexpr int kQB = 64;                // query rows per block
constexpr int kNCHUNK = 4;             // s-dimension split for occupancy
constexpr int kSTILES = (kSQ + 15) / 16;                 // 343
constexpr int kCHT = (kSTILES + kNCHUNK - 1) / kNCHUNK;  // 86 s-tiles per chunk
constexpr int kRP = 2 * kD;            // global row pitch: [hi 768 | lo 768]
constexpr int kLP = kRP + 8;           // LDS row pitch 1544 el = 3088 B:
                                       // lane stride = 772 DW = 4 banks (mod 64)
                                       // -> 16 lanes x 4 banks tile all 64 banks

// ---------------------------------------------------------------------------
// Kernel 1: mean over L, L2-normalize, split fp32 -> bf16 hi + bf16 lo,
// stored interleaved per row: [hi 0..767 | lo 0..767] (row pitch 1536).
// One 256-thread block per row (768 = 3 * 256 elements).
// ---------------------------------------------------------------------------
__global__ void __launch_bounds__(256)
prep_kernel(const float* __restrict__ qf, const float* __restrict__ sf,
            __bf16* __restrict__ xq, __bf16* __restrict__ xs) {
  const int row = blockIdx.x;
  const int tid = threadIdx.x;

  const float* src;
  size_t lstride;
  __bf16* dst;
  if (row < kNQROWS) {
    const int n = row / kQ, q = row % kQ;
    src = qf + ((size_t)n * kT + (q + 1)) * kD;
    lstride = (size_t)kN * kT * kD;
    dst = xq + (size_t)row * kRP;
  } else {
    const int r = row - kNQROWS;
    const int n = r / kSQ, s = r % kSQ;
    const int b = n * kS + s / kQ;       // row in the (N*S) batch dim
    const int t = 1 + s % kQ;            // token index (skip CLS)
    src = sf + ((size_t)b * kT + t) * kD;
    lstride = (size_t)(kN * kS) * kT * kD;
    dst = xs + (size_t)r * kRP;
  }

  float v[3];
  float ss = 0.0f;
#pragma unroll
  for (int j = 0; j < 3; ++j) {
    const int e = tid + j * 256;
    float a = src[e] + src[e + lstride] + src[e + 2 * lstride];
    a *= (1.0f / 3.0f);
    v[j] = a;
    ss += a * a;
  }
  // wave32 reduce, then cross-wave via LDS
#pragma unroll
  for (int off = 16; off > 0; off >>= 1) ss += __shfl_xor(ss, off, 32);
  __shared__ float sbuf[8];
  const int wave = tid >> 5, lane = tid & 31;
  if (lane == 0) sbuf[wave] = ss;
  __syncthreads();
  float tot = 0.0f;
#pragma unroll
  for (int w = 0; w < 8; ++w) tot += sbuf[w];

  const float scale = 1.0f / fmaxf(sqrtf(tot), 1e-12f);
#pragma unroll
  for (int j = 0; j < 3; ++j) {
    const int e = tid + j * 256;
    const float y = v[j] * scale;
    const __bf16 h = (__bf16)y;          // RNE to bf16
    dst[e]      = h;
    dst[kD + e] = (__bf16)(y - (float)h);  // residual for hi/lo split
  }
}

// ---------------------------------------------------------------------------
// Kernel 2: sim-max GEMM. Block = 8 waves, 64 q-rows x one s-chunk.
// A-tile (hi|lo interleaved, bank-conflict-free padded pitch) staged in LDS
// (~193 KB of the WGP's 320 KB). Each wave processes TWO 16-column B tiles:
// per 16x16x32 K-step, 16 DS + 8 VMEM loads feed 24 WMMAs (hi*hi+hi*lo+lo*hi
// per tile). Per-row-tile LDS bases are pinned via empty inline asm so every
// ds_load uses base + 16-bit immediate (no per-step address VALU -> no
// WMMA->VALU WAR hazard NOPs).
// ---------------------------------------------------------------------------
__global__ void __launch_bounds__(256, 1)
simmax_kernel(const __bf16* __restrict__ xq, const __bf16* __restrict__ xs,
              float* __restrict__ partial) {
  extern __shared__ __bf16 lds[];
  __bf16* amat = lds;                        // 64 rows * 1544 el = 193 KB
  float*  red  = (float*)(lds + kQB * kLP);  // 8 waves * 64 rows = 2 KB

  const int qb    = blockIdx.x;
  const int bn    = blockIdx.y;
  const int chunk = blockIdx.z;
  const int tid   = threadIdx.x;
  const int wave  = tid >> 5;
  const int lane  = tid & 31;

  // ---- Stage the 64-row interleaved A tile into LDS (16B chunks) ----
  const int U4R = kRP * 2 / 16;  // 192 uint4 of payload per row
  for (int c = tid; c < kQB * U4R; c += 256) {
    const int r = c / U4R, k = c % U4R;
    int grow = qb * kQB + r;
    if (grow > kQ - 1) grow = kQ - 1;      // clamp: duplicate rows, never stored
    ((uint4*)(amat + (size_t)r * kLP))[k] =
        ((const uint4*)(xq + (size_t)(bn * kQ + grow) * kRP))[k];
  }
  __syncthreads();

  const int col   = lane & 15;            // A row within 16-tile / B column
  const int bkoff = (lane >> 4) << 4;     // B: lanes 16-31 hold K+16..K+31
  const int akoff = (lane >> 4) << 4;     // A: lanes 16-31 hold K+16..K+31

  // Per-row-tile lane base offsets into LDS, laundered through empty asm so
  // the compiler keeps 4 independent address registers (defeats rebasing
  // onto one pointer with >16-bit offsets recomputed per k-step).
  unsigned abase[4];
#pragma unroll
  for (int r = 0; r < 4; ++r) {
    unsigned o = (unsigned)((r * 16 + col) * kLP + akoff);
    asm("" : "+v"(o));
    abase[r] = o;
  }

  v8f vmax[4];
#pragma unroll
  for (int r = 0; r < 4; ++r)
#pragma unroll
    for (int j = 0; j < 8; ++j) vmax[r][j] = -3.0e38f;

  const int st_begin = chunk * kCHT;
  const int st_end   = min(st_begin + kCHT, kSTILES);
  for (int st0 = st_begin + 2 * wave; st0 < st_end; st0 += 16) {
    const int st1 = (st0 + 1 < st_end) ? st0 + 1 : st0;  // duplicate-safe tail
    int s0 = st0 * 16 + col; if (s0 > kSQ - 1) s0 = kSQ - 1;
    int s1 = st1 * 16 + col; if (s1 > kSQ - 1) s1 = kSQ - 1;
    const __bf16* brow0 = xs + (size_t)(bn * kSQ + s0) * kRP + bkoff;
    const __bf16* brow1 = xs + (size_t)(bn * kSQ + s1) * kRP + bkoff;

    v8f acc0[4], acc1[4];
#pragma unroll
    for (int r = 0; r < 4; ++r)
#pragma unroll
      for (int j = 0; j < 8; ++j) { acc0[r][j] = 0.0f; acc1[r][j] = 0.0f; }

    for (int kb = 0; kb < kD; kb += 32) {
      // B fragments: 32 contiguous bytes per lane, hi then lo (+1536 B)
      const v16bf bh0 = *(const v16bf*)(brow0 + kb);
      const v16bf bl0 = *(const v16bf*)(brow0 + kD + kb);
      const v16bf bh1 = *(const v16bf*)(brow1 + kb);
      const v16bf bl1 = *(const v16bf*)(brow1 + kD + kb);
#pragma unroll
      for (int r = 0; r < 4; ++r) {
        const __bf16* ar = amat + abase[r];
        // A fragments: two 16B DS reads each for hi and lo halves
        const v8bf ah0 = *(const v8bf*)(ar + kb);
        const v8bf ah1 = *(const v8bf*)(ar + kb + 8);
        const v8bf al0 = *(const v8bf*)(ar + kD + kb);
        const v8bf al1 = *(const v8bf*)(ar + kD + kb + 8);
        const v16bf ah = __builtin_shufflevector(ah0, ah1,
            0,1,2,3,4,5,6,7,8,9,10,11,12,13,14,15);
        const v16bf al = __builtin_shufflevector(al0, al1,
            0,1,2,3,4,5,6,7,8,9,10,11,12,13,14,15);
        acc0[r] = __builtin_amdgcn_wmma_f32_16x16x32_bf16(
            false, ah, false, bh0, (short)0, acc0[r], false, false);
        acc1[r] = __builtin_amdgcn_wmma_f32_16x16x32_bf16(
            false, ah, false, bh1, (short)0, acc1[r], false, false);
        acc0[r] = __builtin_amdgcn_wmma_f32_16x16x32_bf16(
            false, ah, false, bl0, (short)0, acc0[r], false, false);
        acc1[r] = __builtin_amdgcn_wmma_f32_16x16x32_bf16(
            false, ah, false, bl1, (short)0, acc1[r], false, false);
        acc0[r] = __builtin_amdgcn_wmma_f32_16x16x32_bf16(
            false, al, false, bh0, (short)0, acc0[r], false, false);
        acc1[r] = __builtin_amdgcn_wmma_f32_16x16x32_bf16(
            false, al, false, bh1, (short)0, acc1[r], false, false);
      }
    }
#pragma unroll
    for (int r = 0; r < 4; ++r)
#pragma unroll
      for (int j = 0; j < 8; ++j)
        vmax[r][j] = fmaxf(vmax[r][j], fmaxf(acc0[r][j], acc1[r][j]));
  }

  // Reduce over the 16 columns held by each 16-lane half (C layout:
  // VGPR j, lanes 0-15 -> M=j, lanes 16-31 -> M=j+8, N = lane&15).
#pragma unroll
  for (int r = 0; r < 4; ++r)
#pragma unroll
    for (int m = 1; m < 16; m <<= 1)
#pragma unroll
      for (int j = 0; j < 8; ++j)
        vmax[r][j] = fmaxf(vmax[r][j], __shfl_xor(vmax[r][j], m, 32));

  if ((lane & 15) == 0) {
    const int half8 = (lane >> 4) << 3;
#pragma unroll
    for (int r = 0; r < 4; ++r)
#pragma unroll
      for (int j = 0; j < 8; ++j)
        red[wave * kQB + r * 16 + half8 + j] = vmax[r][j];
  }
  __syncthreads();

  if (tid < kQB) {
    float m = red[tid];
#pragma unroll
    for (int w = 1; w < 8; ++w) m = fmaxf(m, red[w * kQB + tid]);
    const int q = qb * kQB + tid;
    if (q < kQ)
      partial[(size_t)(bn * kQ + q) * kNCHUNK + chunk] = m;
  }
}

// ---------------------------------------------------------------------------
// Kernel 3: combine chunk maxima, dist = 1 - max(sim).
// ---------------------------------------------------------------------------
__global__ void __launch_bounds__(256)
finalize_kernel(const float* __restrict__ partial, float* __restrict__ out) {
  const int t = blockIdx.x * blockDim.x + threadIdx.x;
  if (t < kNQROWS) {
    float m = partial[(size_t)t * kNCHUNK];
#pragma unroll
    for (int c = 1; c < kNCHUNK; ++c)
      m = fmaxf(m, partial[(size_t)t * kNCHUNK + c]);
    out[t] = 1.0f - m;
  }
}

// ---------------------------------------------------------------------------
extern "C" void kernel_launch(void* const* d_in, const int* in_sizes, int n_in,
                              void* d_out, int out_size, void* d_ws, size_t ws_size,
                              hipStream_t stream) {
  const float* qf = (const float*)d_in[0];   // (3, 8, 1370, 768) fp32
  const float* sf = (const float*)d_in[1];   // (3, 32, 1370, 768) fp32

  char* ws = (char*)d_ws;
  size_t off = 0;
  __bf16* xq = (__bf16*)(ws + off); off += (size_t)kNQROWS * kRP * sizeof(__bf16);
  __bf16* xs = (__bf16*)(ws + off); off += (size_t)kNSROWS * kRP * sizeof(__bf16);
  float* partial = (float*)(ws + off); off += (size_t)kNQROWS * kNCHUNK * sizeof(float);
  (void)ws_size; (void)in_sizes; (void)n_in; (void)out_size;

  // 1) mean + l2norm + bf16 hi/lo split (interleaved rows)
  prep_kernel<<<kNQROWS + kNSROWS, 256, 0, stream>>>(qf, sf, xq, xs);

  // 2) WMMA sim-max: grid (q-tiles, images, s-chunks), ~193KB+2KB dynamic LDS
  const size_t lds_bytes = (size_t)kQB * kLP * sizeof(__bf16) + 8 * kQB * sizeof(float);
  simmax_kernel<<<dim3((kQ + kQB - 1) / kQB, kN, kNCHUNK), 256, lds_bytes, stream>>>(
      xq, xs, partial);

  // 3) dist = 1 - max
  finalize_kernel<<<(kNQROWS + 255) / 256, 256, 0, stream>>>(partial, (float*)d_out);
}
